// Encoder_7_50474455662812
// MI455X (gfx1250) — matrix operand order repaired and checked by reading, not compiled, not run on hardware
//
#include <hip/hip_runtime.h>
#include <hip/hip_bf16.h>
#include <math.h>

// ---------------------------------------------------------------------------
// MI455X (gfx1250) encoder. Conv layers are compute-bound (~75 GFLOP vs a few
// hundred MB of traffic at 23.3 TB/s) -> f16 WMMA (v_wmma_f32_16x16x32_f16),
// f32 accumulate, wave32. Weights pre-transposed/padded to f16 [Cout][Kp];
// all WMMA fragments are contiguous 32B runs in LDS -> ds_load_b128 pairs.
// ---------------------------------------------------------------------------

typedef __attribute__((ext_vector_type(16))) _Float16 v16h;
typedef __attribute__((ext_vector_type(8)))  _Float16 v8h;
typedef __attribute__((ext_vector_type(8)))  float    v8f;

#define TPAD   192            // MAX_LEN_PAD
#define NB     64             // batch
#define MROWS  (NB * TPAD)    // 12288 GEMM rows (tiles never cross batch: 192%16==0)
#define DENC   512
#define DENC3  256
#define NECK   8
#define NECK3  32
#define NSEG   7              // MAX_NUM_SEG
#define SEGL2  64             // MAX_SEG*2

__device__ __forceinline__ unsigned hash_u32(unsigned x) {
    x ^= x >> 16; x *= 0x7feb352du;
    x ^= x >> 15; x *= 0x846ca68bu;
    x ^= x >> 16; return x;
}
__device__ __forceinline__ float sigmoidf_(float x) { return 1.0f / (1.0f + __expf(-x)); }

// A-chunk layout: element (row r, k) -> As[r*32 + k] so that each
// lane's 16 A-halfs (K = hi*16..hi*16+15) are one contiguous 32B run.
__device__ __forceinline__ int aperm(int k) {
    return k;
}

// ---------------------------------------------------------------------------
// Weight prep: f32 [K, Cout] -> f16 transposed+padded [Cout, Kp], Kp = ceil32(K).
// Runs once per launch; result stays resident in L2 (192 MB).
// ---------------------------------------------------------------------------
__global__ void cvt_transpose_f16(const float* __restrict__ in,
                                  _Float16* __restrict__ out,
                                  int K, int Cout, int Kp)
{
    const int k = blockIdx.x * 256 + threadIdx.x;
    const int n = blockIdx.y;
    if (k < Kp)
        out[(size_t)n * Kp + k] = (k < K) ? (_Float16)in[(size_t)k * Cout + n]
                                          : (_Float16)0.0f;
}

// ---------------------------------------------------------------------------
// Conv1d(k=5, SAME) as GEMM with on-the-fly im2col. K = 5*Cin, Kp padded.
// Block = 128 threads (4 waves). Block tile 16M x 128N; each wave 16M x 32N
// (two accumulators -> two WMMAs per A fragment).
// ---------------------------------------------------------------------------
__global__ void conv_gemm_wmma(const float* __restrict__ X,
                               const _Float16* __restrict__ W16t, // [Cout][Kp]
                               const float* __restrict__ bias,
                               float* __restrict__ Y,
                               int Cin, int Cout, int Kp, int ldc, int coff)
{
    const int K    = 5 * Cin;
    const int lane = threadIdx.x & 31;
    const int wv   = threadIdx.x >> 5;
    const int m0   = blockIdx.x * 16;
    const int nblk = blockIdx.y * 128;          // block's first output column
    const int nw   = wv * 32;                   // wave's base col inside block tile
    const int b    = m0 / TPAD;
    const int t0   = m0 % TPAD;
    const int rlo  = lane & 15;
    const int hi   = lane >> 4;

    __shared__ __align__(32) _Float16 As[16 * 32];   // permuted A chunk
    __shared__ __align__(32) _Float16 Bs[128 * 32];  // Bs[col][k] column-major

    v8f acc0, acc1;
    const float b0i = bias[nblk + nw + rlo];
    const float b1i = bias[nblk + nw + 16 + rlo];
    #pragma unroll
    for (int r = 0; r < 8; ++r) { acc0[r] = b0i; acc1[r] = b1i; }

    for (int kb = 0; kb < Kp; kb += 32) {
        const int kw0 = kb / Cin;               // scalar division, once per chunk
        const int bnd = (kw0 + 1) * Cin;
        __syncthreads();

        // stage A chunk (im2col + f32->f16), permuted; pure int index math
        #pragma unroll
        for (int e = threadIdx.x; e < 16 * 32; e += 128) {
            const int r  = e >> 5;
            const int k  = e & 31;
            const int kk = kb + k;
            float v = 0.0f;
            if (kk < K) {
                const int kw = (kk >= bnd) ? (kw0 + 1) : kw0;
                const int ci = kk - kw * Cin;
                const int ts = t0 + r + kw - 2;
                if (ts >= 0 && ts < TPAD)
                    v = X[(b * TPAD + ts) * ldc + coff + ci];
            }
            As[r * 32 + aperm(k)] = (_Float16)v;
        }
        // stage B chunk: thread tid -> column nblk+tid, 64 contiguous bytes
        {
            const v8h* src = (const v8h*)(W16t + (size_t)(nblk + threadIdx.x) * Kp + kb);
            v8h* dst = (v8h*)(Bs + threadIdx.x * 32);
            dst[0] = src[0]; dst[1] = src[1]; dst[2] = src[2]; dst[3] = src[3];
        }
        if (kb + 32 < Kp)   // prefetch next weight chunk (global_prefetch_b8)
            __builtin_prefetch(W16t + (size_t)(nblk + threadIdx.x) * Kp + kb + 32, 0, 1);
        __syncthreads();

        // fragments: contiguous 32B LDS runs -> 2x ds_load_b128 each
        const v16h a   = *(const v16h*)(As + rlo * 32 + hi * 16);
        const v16h bf0 = *(const v16h*)(Bs + (nw + rlo) * 32 + hi * 16);
        const v16h bf1 = *(const v16h*)(Bs + (nw + 16 + rlo) * 32 + hi * 16);

        acc0 = __builtin_amdgcn_wmma_f32_16x16x32_f16(false, a, false, bf0,
                                                      (short)0, acc0, false, false);
        acc1 = __builtin_amdgcn_wmma_f32_16x16x32_f16(false, a, false, bf1,
                                                      (short)0, acc1, false, false);
    }

    // D layout: lanes 0-15 -> M=m0+r; lanes 16-31 -> M=m0+8+r; N = base+(lane&15)
    const int nc = nblk + nw + rlo;
    #pragma unroll
    for (int r = 0; r < 8; ++r) {
        const int m = m0 + (hi ? 8 + r : r);
        Y[(size_t)m * Cout + nc]      = acc0[r];
        Y[(size_t)m * Cout + nc + 16] = acc1[r];
    }
}

// ---------------------------------------------------------------------------
// Plain GEMM (LSTM input projection): Y[M,N] = A[M,Kdim] * W + bias, with
// W16t f16 transposed [N][Kdim] (Kdim = 512/256, already multiples of 32).
// Block = 64 threads (2 waves); block tile 16M x 32N; wave -> one 16N tile.
// ---------------------------------------------------------------------------
__global__ void gemm_wmma(const float* __restrict__ A,
                          const _Float16* __restrict__ W16t, // [N][Kdim]
                          const float* __restrict__ bias,
                          float* __restrict__ Y,
                          int Kdim, int N)
{
    const int lane = threadIdx.x & 31;
    const int wv   = threadIdx.x >> 5;
    const int m0   = blockIdx.x * 16;
    const int nblk = blockIdx.y * 32;
    const int rlo  = lane & 15;
    const int hi   = lane >> 4;

    __shared__ __align__(32) _Float16 As[16 * 32];
    __shared__ __align__(32) _Float16 Bs[32 * 32];   // Bs[col][k]

    v8f acc;
    const float bcol = bias[nblk + wv * 16 + rlo];
    #pragma unroll
    for (int r = 0; r < 8; ++r) acc[r] = bcol;

    for (int kb = 0; kb < Kdim; kb += 32) {
        __syncthreads();
        #pragma unroll
        for (int e = threadIdx.x; e < 16 * 32; e += 64) {
            const int r = e >> 5;
            const int k = e & 31;
            As[r * 32 + aperm(k)] = (_Float16)A[(m0 + r) * Kdim + kb + k];
        }
        {   // 32 columns x 32 K halfs; thread -> (col, 16-half section)
            const int col = threadIdx.x & 31;
            const int sec = threadIdx.x >> 5;    // 0 or 1
            const v8h* src = (const v8h*)(W16t + (size_t)(nblk + col) * Kdim + kb + sec * 16);
            v8h* dst = (v8h*)(Bs + col * 32 + sec * 16);
            dst[0] = src[0]; dst[1] = src[1];
        }
        __syncthreads();

        const v16h a  = *(const v16h*)(As + rlo * 32 + hi * 16);
        const v16h bf = *(const v16h*)(Bs + (wv * 16 + rlo) * 32 + hi * 16);
        acc = __builtin_amdgcn_wmma_f32_16x16x32_f16(false, a, false, bf,
                                                     (short)0, acc, false, false);
    }

    const int nc = nblk + wv * 16 + rlo;
    #pragma unroll
    for (int r = 0; r < 8; ++r) {
        const int m = m0 + (hi ? 8 + r : r);
        Y[(size_t)m * N + nc] = acc[r];
    }
}

// ---------------------------------------------------------------------------
// GroupNorm(16 ch/group) + tanh. One block per (batch, group); reduce T*16.
// ---------------------------------------------------------------------------
__global__ void gn_tanh(const float* __restrict__ X,
                        const float* __restrict__ gamma,
                        const float* __restrict__ beta,
                        float* __restrict__ Y, int C)
{
    const int b  = blockIdx.x;
    const int c0 = blockIdx.y * 16;
    __shared__ float ssum[256], ssq[256];

    float s = 0.0f, sq = 0.0f;
    for (int e = threadIdx.x; e < TPAD * 16; e += blockDim.x) {
        const int t = e >> 4;
        const int c = c0 + (e & 15);
        const float v = X[((size_t)b * TPAD + t) * C + c];
        s += v; sq += v * v;
    }
    ssum[threadIdx.x] = s; ssq[threadIdx.x] = sq;
    __syncthreads();
    for (int st = 128; st > 0; st >>= 1) {
        if (threadIdx.x < st) {
            ssum[threadIdx.x] += ssum[threadIdx.x + st];
            ssq[threadIdx.x]  += ssq[threadIdx.x + st];
        }
        __syncthreads();
    }
    const float inv = 1.0f / (float)(TPAD * 16);
    const float mu  = ssum[0] * inv;
    const float var = ssq[0] * inv - mu * mu;
    const float rs  = rsqrtf(var + 1e-3f);

    for (int e = threadIdx.x; e < TPAD * 16; e += blockDim.x) {
        const int t = e >> 4;
        const int c = c0 + (e & 15);
        const size_t idx = ((size_t)b * TPAD + t) * C + c;
        Y[idx] = tanhf((X[idx] - mu) * rs * gamma[c] + beta[c]);
    }
}

// ---------------------------------------------------------------------------
// Random-resampling (interp_lnr equivalent, hash-based PRNG). One block/batch.
// ---------------------------------------------------------------------------
__global__ void interp_resample(const float* __restrict__ Xin,  // [MROWS,512]
                                const float* __restrict__ Fin,  // [MROWS,256]
                                float* __restrict__ Xout,
                                float* __restrict__ Fout,
                                int layer)
{
    const int b = blockIdx.x;
    __shared__ int   srci[TPAD];
    __shared__ float lamb[TPAD];
    __shared__ int   cnt_s;

    if (threadIdx.x == 0) {
        int cnt = 0, off = 0;
        for (int sgi = 0; sgi < NSEG; ++sgi) {
            const unsigned r = (unsigned)(b * NSEG + sgi);
            const unsigned base = (unsigned)layer * 100003u + r * 2u + 7u;
            const float scale = (float)hash_u32(base) * (1.0f / 4294967296.0f) + 0.5f;
            const int   seg   = 19 + (int)(hash_u32(base + 1u) % 13u); // [19,32)
            for (int j = 0; j < SEGL2 && cnt < TPAD; ++j) {
                const float idx = (float)j / scale;
                const float fl  = floorf(idx);
                if (fl < (float)seg - 1.0f) {
                    const float io = fl + (float)off;
                    if (io < (float)(TPAD - 1)) {
                        srci[cnt] = (int)io;
                        lamb[cnt] = idx - fl;
                        ++cnt;
                    }
                }
            }
            off += seg;
        }
        cnt_s = cnt;
    }
    __syncthreads();
    const int cnt = cnt_s;

    for (int e = threadIdx.x; e < TPAD * (DENC + DENC3); e += blockDim.x) {
        const int slot = e / (DENC + DENC3);
        const int c    = e - slot * (DENC + DENC3);
        float y = 0.0f;
        if (slot < cnt) {
            int i0 = srci[slot]; if (i0 < 0) i0 = 0; if (i0 > TPAD - 1) i0 = TPAD - 1;
            int i1 = i0 + 1;     if (i1 > TPAD - 1) i1 = TPAD - 1;
            const float l = lamb[slot];
            if (c < DENC) {
                y = (1.0f - l) * Xin[(b * TPAD + i0) * DENC + c]
                  +          l * Xin[(b * TPAD + i1) * DENC + c];
            } else {
                const int cf = c - DENC;
                y = (1.0f - l) * Fin[(b * TPAD + i0) * DENC3 + cf]
                  +          l * Fin[(b * TPAD + i1) * DENC3 + cf];
            }
        }
        if (c < DENC) Xout[(b * TPAD + slot) * DENC  + c]          = y;
        else          Fout[(b * TPAD + slot) * DENC3 + (c - DENC)] = y;
    }
}

// ---------------------------------------------------------------------------
// LSTM recurrence. One wave per batch element; h/c state in LDS.
// ---------------------------------------------------------------------------
__global__ void lstm_rec(const float* __restrict__ xp,
                         const float* __restrict__ wh,   // [H, 4H]
                         float* __restrict__ out,
                         int H, int outC, int coff, int reverse)
{
    const int b    = blockIdx.x;
    const int lane = threadIdx.x;       // 0..31
    const int Z    = 4 * H;
    const int per  = Z / 32;            // 1 (H=8) or 4 (H=32)

    __shared__ float h[32], c[32], zbuf[128];
    if (lane < H) { h[lane] = 0.0f; c[lane] = 0.0f; }
    __syncthreads();

    for (int step = 0; step < TPAD; ++step) {
        const int t = reverse ? (TPAD - 1 - step) : step;
        float zn[4];
        for (int q = 0; q < per; ++q) {
            const int j = lane + q * 32;
            float z = xp[(b * TPAD + t) * Z + j];
            for (int k = 0; k < H; ++k) z += h[k] * wh[k * Z + j];
            zn[q] = z;
        }
        for (int q = 0; q < per; ++q) zbuf[lane + q * 32] = zn[q];
        __syncthreads();
        if (lane < H) {
            const float i_ = zbuf[lane];
            const float f_ = zbuf[H + lane];
            const float g_ = zbuf[2 * H + lane];
            const float o_ = zbuf[3 * H + lane];
            const float cn = sigmoidf_(f_) * c[lane] + sigmoidf_(i_) * tanhf(g_);
            c[lane] = cn;
            const float hn = sigmoidf_(o_) * tanhf(cn);
            h[lane] = hn;
            out[(b * TPAD + t) * outC + coff + lane] = hn;
        }
        __syncthreads();
    }
}

// ---------------------------------------------------------------------------
// Final strided gather: codes_x (64,24,16) ++ codes_f0 (64,24,64) flat.
// ---------------------------------------------------------------------------
__global__ void gather_codes(const float* __restrict__ xh,   // [MROWS,16]
                             const float* __restrict__ fh,   // [MROWS,64]
                             float* __restrict__ out)
{
    const int idx = blockIdx.x * blockDim.x + threadIdx.x;
    const int totx = NB * 24 * 16;
    const int totf = NB * 24 * 64;
    if (idx < totx) {
        const int c = idx & 15;
        const int s = (idx >> 4) % 24;
        const int b = idx / (24 * 16);
        const int t = (c < NECK) ? (s * 8 + 7) : (s * 8);
        out[idx] = xh[(b * TPAD + t) * 16 + c];
    } else if (idx < totx + totf) {
        const int j = idx - totx;
        const int c = j & 63;
        const int s = (j >> 6) % 24;
        const int b = j / (24 * 64);
        const int t = (c < NECK3) ? (s * 8 + 7) : (s * 8);
        out[idx] = fh[(b * TPAD + t) * 64 + c];
    }
}

// ---------------------------------------------------------------------------
extern "C" void kernel_launch(void* const* d_in, const int* in_sizes, int n_in,
                              void* d_out, int out_size, void* d_ws, size_t ws_size,
                              hipStream_t stream)
{
    (void)in_sizes; (void)n_in; (void)out_size; (void)ws_size;
    const float* x_f0 = (const float*)d_in[0];

    // ---- workspace layout: f32 region first, then f16 weight region ----
    float* ws = (float*)d_ws;
    size_t o = 0;
    float* Xa = ws + o; o += (size_t)MROWS * DENC;
    float* Xb = ws + o; o += (size_t)MROWS * DENC;
    float* Fa = ws + o; o += (size_t)MROWS * DENC3;
    float* Fb = ws + o; o += (size_t)MROWS * DENC3;
    float* xp1f = ws + o; o += (size_t)MROWS * 4 * NECK;
    float* xp1b = ws + o; o += (size_t)MROWS * 4 * NECK;
    float* xp2f = ws + o; o += (size_t)MROWS * 4 * NECK3;
    float* xp2b = ws + o; o += (size_t)MROWS * 4 * NECK3;
    float* xh   = ws + o; o += (size_t)MROWS * 2 * NECK;
    float* fh   = ws + o; o += (size_t)MROWS * 2 * NECK3;

    _Float16* hbase = (_Float16*)(ws + o);
    size_t ho = 0;
    const int cin1[3] = {80, 512, 512};
    const int cin2[3] = {257, 256, 256};
    int kp1[3], kp2[3];
    _Float16* w1h[3]; _Float16* w2h[3];
    for (int i = 0; i < 3; ++i) {
        kp1[i] = (5 * cin1[i] + 31) & ~31;
        w1h[i] = hbase + ho; ho += (size_t)DENC * kp1[i];
    }
    for (int i = 0; i < 3; ++i) {
        kp2[i] = (5 * cin2[i] + 31) & ~31;
        w2h[i] = hbase + ho; ho += (size_t)DENC3 * kp2[i];
    }
    _Float16* wx1fh = hbase + ho; ho += (size_t)(4 * NECK)  * DENC;
    _Float16* wx1bh = hbase + ho; ho += (size_t)(4 * NECK)  * DENC;
    _Float16* wx2fh = hbase + ho; ho += (size_t)(4 * NECK3) * DENC3;
    _Float16* wx2bh = hbase + ho; ho += (size_t)(4 * NECK3) * DENC3;

    // ---- weight prep: f32 [K,Cout] -> f16 transposed [Cout,Kp] ----
    for (int i = 0; i < 3; ++i) {
        cvt_transpose_f16<<<dim3((kp1[i] + 255) / 256, DENC),  256, 0, stream>>>(
            (const float*)d_in[1 + i * 8 + 0], w1h[i], 5 * cin1[i], DENC,  kp1[i]);
        cvt_transpose_f16<<<dim3((kp2[i] + 255) / 256, DENC3), 256, 0, stream>>>(
            (const float*)d_in[1 + i * 8 + 4], w2h[i], 5 * cin2[i], DENC3, kp2[i]);
    }
    cvt_transpose_f16<<<dim3(2, 4 * NECK),  256, 0, stream>>>((const float*)d_in[25], wx1fh, DENC,  4 * NECK,  DENC);
    cvt_transpose_f16<<<dim3(2, 4 * NECK),  256, 0, stream>>>((const float*)d_in[31], wx1bh, DENC,  4 * NECK,  DENC);
    cvt_transpose_f16<<<dim3(1, 4 * NECK3), 256, 0, stream>>>((const float*)d_in[28], wx2fh, DENC3, 4 * NECK3, DENC3);
    cvt_transpose_f16<<<dim3(1, 4 * NECK3), 256, 0, stream>>>((const float*)d_in[34], wx2bh, DENC3, 4 * NECK3, DENC3);

    const int mt = MROWS / 16;   // 768 M tiles

    const float* srcX = x_f0; int cinX = 80,  ldX = 337, coX = 0;
    const float* srcF = x_f0; int cinF = 257, ldF = 337, coF = 80;

    for (int i = 0; i < 3; ++i) {
        const int base = 1 + i * 8;
        const float* c1b = (const float*)d_in[base + 1];
        const float* g1g = (const float*)d_in[base + 2];
        const float* g1e = (const float*)d_in[base + 3];
        const float* c2b = (const float*)d_in[base + 5];
        const float* g2g = (const float*)d_in[base + 6];
        const float* g2e = (const float*)d_in[base + 7];

        // conv (WMMA GEMM): src -> Xa / Fa   (block tile = 16M x 128N)
        conv_gemm_wmma<<<dim3(mt, DENC  / 128), 128, 0, stream>>>(srcX, w1h[i], c1b, Xa, cinX, DENC,  kp1[i], ldX, coX);
        conv_gemm_wmma<<<dim3(mt, DENC3 / 128), 128, 0, stream>>>(srcF, w2h[i], c2b, Fa, cinF, DENC3, kp2[i], ldF, coF);
        // GN + tanh: Xa -> Xb, Fa -> Fb
        gn_tanh<<<dim3(NB, DENC  / 16), 256, 0, stream>>>(Xa, g1g, g1e, Xb, DENC);
        gn_tanh<<<dim3(NB, DENC3 / 16), 256, 0, stream>>>(Fa, g2g, g2e, Fb, DENC3);
        // random resample: (Xb,Fb) -> (Xa,Fa)
        interp_resample<<<NB, 256, 0, stream>>>(Xb, Fb, Xa, Fa, i);

        srcX = Xa; cinX = DENC;  ldX = DENC;  coX = 0;
        srcF = Fa; cinF = DENC3; ldF = DENC3; coF = 0;
    }

    // LSTM input projections (WMMA GEMM), then recurrences
    const float* l1whf = (const float*)d_in[26];
    const float* l1bf  = (const float*)d_in[27];
    const float* l2whf = (const float*)d_in[29];
    const float* l2bf  = (const float*)d_in[30];
    const float* l1whb = (const float*)d_in[32];
    const float* l1bb  = (const float*)d_in[33];
    const float* l2whb = (const float*)d_in[35];
    const float* l2bb  = (const float*)d_in[36];

    gemm_wmma<<<dim3(mt, 1), 64, 0, stream>>>(Xa, wx1fh, l1bf, xp1f, DENC,  4 * NECK);
    gemm_wmma<<<dim3(mt, 1), 64, 0, stream>>>(Xa, wx1bh, l1bb, xp1b, DENC,  4 * NECK);
    gemm_wmma<<<dim3(mt, 4), 64, 0, stream>>>(Fa, wx2fh, l2bf, xp2f, DENC3, 4 * NECK3);
    gemm_wmma<<<dim3(mt, 4), 64, 0, stream>>>(Fa, wx2bh, l2bb, xp2b, DENC3, 4 * NECK3);

    lstm_rec<<<NB, 32, 0, stream>>>(xp1f, l1whf, xh, NECK,  2 * NECK,  0,     0);
    lstm_rec<<<NB, 32, 0, stream>>>(xp1b, l1whb, xh, NECK,  2 * NECK,  NECK,  1);
    lstm_rec<<<NB, 32, 0, stream>>>(xp2f, l2whf, fh, NECK3, 2 * NECK3, 0,     0);
    lstm_rec<<<NB, 32, 0, stream>>>(xp2b, l2whb, fh, NECK3, 2 * NECK3, NECK3, 1);

    const int tot = NB * 24 * 16 + NB * 24 * 64;
    gather_codes<<<(tot + 255) / 256, 256, 0, stream>>>(xh, fh, (float*)d_out);
}
